// GATom_73005854097574
// MI455X (gfx1250) — matrix operand order repaired and checked by reading, not compiled, not run on hardware
//
#include <hip/hip_runtime.h>

// ---------------------------------------------------------------------------
// GATv2 GNN pipeline for MI455X (gfx1250, wave32).
// Heavy GEMMs use v_wmma_f32_16x16x32_bf16 with b128 fragment loads
// (weights pre-transposed to [N,K] so every fragment is contiguous).
// Segment softmax via L2-resident float atomics. Workspace ~220 MB.
// ---------------------------------------------------------------------------

typedef __bf16 bf16_t;
typedef bf16_t v16bf __attribute__((ext_vector_type(16)));
typedef bf16_t v8bf  __attribute__((ext_vector_type(8)));
typedef float  v8f   __attribute__((ext_vector_type(8)));

#define DEV __device__ __forceinline__

DEV bf16_t f2bf(float f) {
  unsigned u = __float_as_uint(f);
  unsigned r = (u + 0x7fffu + ((u >> 16) & 1u)) >> 16;   // RNE truncate to bf16
  unsigned short s = (unsigned short)r;
  return __builtin_bit_cast(bf16_t, s);
}
DEV float siluf(float t) { return t / (1.f + __expf(-t)); }
DEV float eluf(float t)  { return t > 0.f ? t : (__expf(t) - 1.f); }
DEV float lreluf(float t){ return t > 0.f ? t : 0.01f * t; }
// order-preserving float <-> uint encoding for atomicMax over floats
DEV unsigned fenc(float f) { unsigned u = __float_as_uint(f); return (u & 0x80000000u) ? ~u : (u | 0x80000000u); }
DEV float fdec(unsigned u) { return (u & 0x80000000u) ? __uint_as_float(u & 0x7fffffffu) : __uint_as_float(~u); }
#define ENC_NEG_INF 0x007FFFFFu  // fenc(-inf)

// --- WMMA bf16 fragment loaders (CDNA5 ISA layouts) --------------
// A 16x32 (MxK), row-major source. Lane layout = two contiguous 8-elem runs:
//   lanes 0-15 : M=lane,    K in [0,8)  and [16,24)
//   lanes 16-31: M=lane-16, K in [8,16) and [24,32)
DEV v16bf load_frag_a(const bf16_t* __restrict__ A, int ld, int row0, int k0, int lane) {
  const bf16_t* p = A + (size_t)(row0 + (lane & 15)) * (size_t)ld + k0 + ((lane < 16) ? 0 : 8);
  v8bf lo = *(const v8bf*)p;          // K = khalf + 0..7
  v8bf hi = *(const v8bf*)(p + 16);   // K = khalf + 16..23
  return __builtin_shufflevector(lo, hi, 0,1,2,3,4,5,6,7,8,9,10,11,12,13,14,15);
}
// B 32x16 (KxN) fragment, from weight stored TRANSPOSED as Wt[N, Ktot]:
//   lanes 0-15 : N=lane,    K = k0 + [0,8) and [16,24)
//   lanes 16-31: N=lane-16, K = k0 + [8,16) and [24,32)
DEV v16bf load_frag_bt(const bf16_t* __restrict__ Wt, int ldk, int k0, int col0, int lane) {
  const bf16_t* p = Wt + (size_t)(col0 + (lane & 15)) * (size_t)ldk + k0 + ((lane < 16) ? 0 : 8);
  v8bf lo = *(const v8bf*)p;
  v8bf hi = *(const v8bf*)(p + 16);
  return __builtin_shufflevector(lo, hi, 0,1,2,3,4,5,6,7,8,9,10,11,12,13,14,15);
}
DEV v8f wmma_bf16(v16bf a, v16bf b, v8f c) {
  return __builtin_amdgcn_wmma_f32_16x16x32_bf16(false, a, false, b, (short)0, c, false, false);
}
// C/D 16x16 f32: VGPR r -> row rbase+r (rbase = lane<16 ? 0 : 8), col = lane&15.

// ---------------------------------------------------------------------------
// Utility kernels
// ---------------------------------------------------------------------------
__global__ void k_cvt_bf16(const float* __restrict__ s, bf16_t* __restrict__ d, int n) {
  for (int i = blockIdx.x * blockDim.x + threadIdx.x; i < n; i += gridDim.x * blockDim.x)
    d[i] = f2bf(s[i]);
}
// batched transpose-convert: src[b][K][N] (f32, row-major) -> dst[b][N][K] (bf16)
__global__ void k_cvt_t(const float* __restrict__ src, bf16_t* __restrict__ dst,
                        int K, int N, int total) {
  int i = blockIdx.x * blockDim.x + threadIdx.x;
  if (i >= total) return;
  int kn = K * N;
  int b = i / kn, j = i - b * kn;
  int n = j / K, k = j - n * K;
  dst[i] = f2bf(src[(size_t)b * kn + (size_t)k * N + n]);
}
__global__ void k_zero(float* __restrict__ p, int n) {
  int i = blockIdx.x * blockDim.x + threadIdx.x;
  if (i < n) p[i] = 0.f;
}
// init per-segment state: acc[n_seg*64]=0, max=-inf, den=0 (launch n_seg*64 threads)
__global__ void k_init_seg(unsigned* __restrict__ mx, float* __restrict__ den,
                           float* __restrict__ acc, int n_seg) {
  int i = blockIdx.x * blockDim.x + threadIdx.x;
  if (i < n_seg * 64) acc[i] = 0.f;
  if (i < n_seg) { mx[i] = ENC_NEG_INF; den[i] = 0.f; }
}

// h = silu(x @ W[92,64] + b)   (thread = (node, col))
__global__ void k_pre_node(const float* __restrict__ x, const float* __restrict__ W,
                           const float* __restrict__ b, float* __restrict__ h, int total) {
  int i = blockIdx.x * blockDim.x + threadIdx.x;
  if (i >= total) return;
  int n = i >> 6, c = i & 63;
  const float* xr = x + (size_t)n * 92;
  float acc = b[c];
#pragma unroll 4
  for (int k = 0; k < 92; ++k) acc += xr[k] * W[k * 64 + c];
  h[i] = siluf(acc);
}
// ea_bf = bf16(silu(edge_attr @ W[50,64] + b))   (thread = (edge, col))
__global__ void k_pre_edge(const float* __restrict__ ea, const float* __restrict__ W,
                           const float* __restrict__ b, bf16_t* __restrict__ out, int total) {
  int i = blockIdx.x * blockDim.x + threadIdx.x;
  if (i >= total) return;
  int e = i >> 6, c = i & 63;
  const float* er = ea + (size_t)e * 50;
  float acc = b[c];
#pragma unroll 5
  for (int k = 0; k < 50; ++k) acc += er[k] * W[k * 64 + c];
  out[i] = f2bf(siluf(acc));
}
// hn_bf = bf16(layernorm(h) * g + b)   (thread = row; float4 in, v8bf out)
__global__ void k_layernorm_bf(const float* __restrict__ h, const float* __restrict__ g,
                               const float* __restrict__ b, bf16_t* __restrict__ out, int N) {
  int n = blockIdx.x * blockDim.x + threadIdx.x;
  if (n >= N) return;
  const float4* row = (const float4*)(h + (size_t)n * 64);
  float4 r[16];
#pragma unroll
  for (int k = 0; k < 16; ++k) r[k] = row[k];
  float mu = 0.f;
#pragma unroll
  for (int k = 0; k < 16; ++k) mu += r[k].x + r[k].y + r[k].z + r[k].w;
  mu *= (1.f / 64.f);
  float var = 0.f;
#pragma unroll
  for (int k = 0; k < 16; ++k) {
    float dx = r[k].x - mu, dy = r[k].y - mu, dz = r[k].z - mu, dw = r[k].w - mu;
    var += dx * dx + dy * dy + dz * dz + dw * dw;
  }
  var *= (1.f / 64.f);
  float rs = rsqrtf(var + 1e-5f);
  v8bf* orow = (v8bf*)(out + (size_t)n * 64);
#pragma unroll
  for (int q = 0; q < 8; ++q) {
    v8bf o;
#pragma unroll
    for (int t = 0; t < 8; ++t) {
      int k = q * 8 + t;
      float val = (&r[k >> 2].x)[k & 3];
      o[t] = f2bf((val - mu) * rs * g[k] + b[k]);
    }
    orow[q] = o;
  }
}

// out1 = A@W1t + b1 ; out2 = A@W2t + b2  (A bf16 [M,64]; Wt transposed [64,64])
__global__ void k_gemm64_dual(const bf16_t* __restrict__ A,
                              const bf16_t* __restrict__ W1t, const float* __restrict__ b1,
                              float* __restrict__ out1,
                              const bf16_t* __restrict__ W2t, const float* __restrict__ b2,
                              float* __restrict__ out2, int Mtiles) {
  int wave = blockIdx.x * (blockDim.x >> 5) + (threadIdx.x >> 5);
  int lane = threadIdx.x & 31;
  if (wave >= Mtiles) return;
  int row0 = wave * 16;
  v16bf a0 = load_frag_a(A, 64, row0, 0, lane);
  v16bf a1 = load_frag_a(A, 64, row0, 32, lane);
  int n = lane & 15, rbase = (lane < 16) ? 0 : 8;
#pragma unroll
  for (int nt = 0; nt < 4; ++nt) {
    int col = nt * 16;
    v8f acc1 = {}, acc2 = {};
    acc1 = wmma_bf16(a0, load_frag_bt(W1t, 64, 0, col, lane), acc1);
    acc1 = wmma_bf16(a1, load_frag_bt(W1t, 64, 32, col, lane), acc1);
    acc2 = wmma_bf16(a0, load_frag_bt(W2t, 64, 0, col, lane), acc2);
    acc2 = wmma_bf16(a1, load_frag_bt(W2t, 64, 32, col, lane), acc2);
    float bb1 = b1[col + n], bb2 = b2[col + n];
#pragma unroll
    for (int r = 0; r < 8; ++r) {
      size_t idx = (size_t)(row0 + rbase + r) * 64 + col + n;
      out1[idx] = acc1[r] + bb1;
      out2[idx] = acc2[r] + bb2;
    }
  }
}
// single-output variant
__global__ void k_gemm64(const bf16_t* __restrict__ A, const bf16_t* __restrict__ Wt,
                         const float* __restrict__ bias, float* __restrict__ out, int Mtiles) {
  int wave = blockIdx.x * (blockDim.x >> 5) + (threadIdx.x >> 5);
  int lane = threadIdx.x & 31;
  if (wave >= Mtiles) return;
  int row0 = wave * 16;
  v16bf a0 = load_frag_a(A, 64, row0, 0, lane);
  v16bf a1 = load_frag_a(A, 64, row0, 32, lane);
  int n = lane & 15, rbase = (lane < 16) ? 0 : 8;
#pragma unroll
  for (int nt = 0; nt < 4; ++nt) {
    int col = nt * 16;
    v8f acc = {};
    acc = wmma_bf16(a0, load_frag_bt(Wt, 64, 0, col, lane), acc);
    acc = wmma_bf16(a1, load_frag_bt(Wt, 64, 32, col, lane), acc);
    float bb = bias[col + n];
#pragma unroll
    for (int r = 0; r < 8; ++r)
      out[(size_t)(row0 + rbase + r) * 64 + col + n] = acc[r] + bb;
  }
}

// Fused edge kernel: t = ea@We (WMMA); m = t + xl[src] + xr[dst] + be;
// logit = leakyrelu(m) . att ; store logit, atomicMax per dst.
__global__ void k_edge_logit(const bf16_t* __restrict__ ea, const bf16_t* __restrict__ Wet,
                             const float* __restrict__ be, const float* __restrict__ att,
                             const float* __restrict__ xl, const float* __restrict__ xr,
                             const int* __restrict__ src, const int* __restrict__ dst,
                             float* __restrict__ logits, unsigned* __restrict__ dmax,
                             int Etiles) {
  int wave = blockIdx.x * (blockDim.x >> 5) + (threadIdx.x >> 5);
  int lane = threadIdx.x & 31;
  if (wave >= Etiles) return;
  int row0 = wave * 16;
  __builtin_prefetch(ea + (size_t)(row0 + 256) * 64, 0, 1);  // global_prefetch_b8
  v16bf a0 = load_frag_a(ea, 64, row0, 0, lane);
  v16bf a1 = load_frag_a(ea, 64, row0, 32, lane);
  int n = lane & 15, rbase = (lane < 16) ? 0 : 8;
  int se[8], de[8];
  *(int4*)&se[0] = *(const int4*)(src + row0 + rbase);
  *(int4*)&se[4] = *(const int4*)(src + row0 + rbase + 4);
  *(int4*)&de[0] = *(const int4*)(dst + row0 + rbase);
  *(int4*)&de[4] = *(const int4*)(dst + row0 + rbase + 4);
  float part[8];
#pragma unroll
  for (int r = 0; r < 8; ++r) part[r] = 0.f;
#pragma unroll
  for (int nt = 0; nt < 4; ++nt) {
    int col = nt * 16;
    v8f acc = {};
    acc = wmma_bf16(a0, load_frag_bt(Wet, 64, 0, col, lane), acc);
    acc = wmma_bf16(a1, load_frag_bt(Wet, 64, 32, col, lane), acc);
    float bev = be[col + n], av = att[col + n];
#pragma unroll
    for (int r = 0; r < 8; ++r) {
      float m = acc[r] + xl[(size_t)se[r] * 64 + col + n] + xr[(size_t)de[r] * 64 + col + n] + bev;
      part[r] += lreluf(m) * av;
    }
  }
  // reduce the 16 lanes of each half-wave (offsets < 16 stay within halves)
#pragma unroll
  for (int off = 8; off > 0; off >>= 1) {
#pragma unroll
    for (int r = 0; r < 8; ++r) part[r] += __shfl_xor(part[r], off, 32);
  }
  if (n == 0) {  // lanes 0 (rows 0..7) and 16 (rows 8..15)
#pragma unroll
    for (int r = 0; r < 8; ++r) {
      int e = row0 + rbase + r;
      logits[e] = part[r];
      atomicMax(&dmax[de[r]], fenc(part[r]));
    }
  }
}

// a = exp(logit - max[dst]); den[dst] += a; g[dst,:] += a * xl[src,:]
// thread = (edge, quad-of-4-channels)
__global__ void k_edge_scatter(const float* __restrict__ logits, const unsigned* __restrict__ dmax,
                               const int* __restrict__ src, const int* __restrict__ dst,
                               const float* __restrict__ xl, float* __restrict__ den,
                               float* __restrict__ g, int E) {
  int tid = blockIdx.x * blockDim.x + threadIdx.x;
  if (tid >= E * 16) return;
  int e = tid >> 4, q = tid & 15;
  int d = dst[e], s = src[e];
  float a = __expf(logits[e] - fdec(dmax[d]));
  if (q == 0) atomicAdd(&den[d], a);
  int c0 = q * 4;
  float4 xv = *(const float4*)(xl + (size_t)s * 64 + c0);
  float* gp = g + (size_t)d * 64 + c0;
  atomicAdd(gp + 0, a * xv.x);
  atomicAdd(gp + 1, a * xv.y);
  atomicAdd(gp + 2, a * xv.z);
  atomicAdd(gp + 3, a * xv.w);
}

// g = g / (den + eps) + bias ; also emit bf16   (thread = (seg, col))
__global__ void k_g_final(float* __restrict__ g, const float* __restrict__ den,
                          const float* __restrict__ bias, bf16_t* __restrict__ gbf, int total) {
  int i = blockIdx.x * blockDim.x + threadIdx.x;
  if (i >= total) return;
  int n = i >> 6, c = i & 63;
  float v = g[i] / (den[n] + 1e-16f) + bias[c];
  g[i] = v;
  gbf[i] = f2bf(v);
}

// GLU: p = [A0|A1] @ W[128,128] + b ; out = p[:, :64] * elu(p[:, 64:])
// Wt is the transposed weight [N=128, K=128].
__global__ void k_glu(const bf16_t* __restrict__ A0, const bf16_t* __restrict__ A1,
                      const bf16_t* __restrict__ Wt, const float* __restrict__ b,
                      float* __restrict__ out, int Mtiles) {
  int wave = blockIdx.x * (blockDim.x >> 5) + (threadIdx.x >> 5);
  int lane = threadIdx.x & 31;
  if (wave >= Mtiles) return;
  int row0 = wave * 16;
  v16bf f0 = load_frag_a(A0, 64, row0, 0, lane);
  v16bf f1 = load_frag_a(A0, 64, row0, 32, lane);
  v16bf f2 = load_frag_a(A1, 64, row0, 0, lane);
  v16bf f3 = load_frag_a(A1, 64, row0, 32, lane);
  int n = lane & 15, rbase = (lane < 16) ? 0 : 8;
#pragma unroll
  for (int nt = 0; nt < 4; ++nt) {
    int cv = nt * 16, cg = 64 + nt * 16;
    v8f av = {}, gv = {};
    av = wmma_bf16(f0, load_frag_bt(Wt, 128, 0,  cv, lane), av);
    av = wmma_bf16(f1, load_frag_bt(Wt, 128, 32, cv, lane), av);
    av = wmma_bf16(f2, load_frag_bt(Wt, 128, 64, cv, lane), av);
    av = wmma_bf16(f3, load_frag_bt(Wt, 128, 96, cv, lane), av);
    gv = wmma_bf16(f0, load_frag_bt(Wt, 128, 0,  cg, lane), gv);
    gv = wmma_bf16(f1, load_frag_bt(Wt, 128, 32, cg, lane), gv);
    gv = wmma_bf16(f2, load_frag_bt(Wt, 128, 64, cg, lane), gv);
    gv = wmma_bf16(f3, load_frag_bt(Wt, 128, 96, cg, lane), gv);
    float bv = b[cv + n], bg = b[cg + n];
#pragma unroll
    for (int r = 0; r < 8; ++r) {
      float vv = av[r] + bv;
      float gg = gv[r] + bg;
      out[(size_t)(row0 + rbase + r) * 64 + cv + n] = vv * eluf(gg);
    }
  }
}

// pooled[batch[n], c] += h[n, c]
__global__ void k_pool_sum(const float* __restrict__ h, const int* __restrict__ batch,
                           float* __restrict__ pooled, int total) {
  int i = blockIdx.x * blockDim.x + threadIdx.x;
  if (i >= total) return;
  int n = i >> 6, c = i & 63;
  atomicAdd(&pooled[(size_t)batch[n] * 64 + c], h[i]);
}
__global__ void k_pool_relu(float* __restrict__ pooled, bf16_t* __restrict__ pbf, int total) {
  int i = blockIdx.x * blockDim.x + threadIdx.x;
  if (i >= total) return;
  float v = fmaxf(pooled[i], 0.f);
  pooled[i] = v;
  pbf[i] = f2bf(v);
}

// global super-node attention logits: one thread per node
__global__ void k_gnode_logit(const float* __restrict__ xl2, const float* __restrict__ xr2,
                              const float* __restrict__ att, const int* __restrict__ batch,
                              float* __restrict__ logits2, unsigned* __restrict__ gmax, int N) {
  int n = blockIdx.x * blockDim.x + threadIdx.x;
  if (n >= N) return;
  int b = batch[n];
  const float4* pl = (const float4*)(xl2 + (size_t)n * 64);
  const float4* pr = (const float4*)(xr2 + (size_t)b * 64);
  const float4* pa = (const float4*)att;
  float acc = 0.f;
#pragma unroll 4
  for (int c = 0; c < 16; ++c) {
    float4 l = pl[c], r = pr[c], a = pa[c];
    acc += lreluf(l.x + r.x) * a.x + lreluf(l.y + r.y) * a.y +
           lreluf(l.z + r.z) * a.z + lreluf(l.w + r.w) * a.w;
  }
  logits2[n] = acc;
  atomicMax(&gmax[b], fenc(acc));
}
__global__ void k_gscatter(const float* __restrict__ logits2, const unsigned* __restrict__ gmax,
                           const int* __restrict__ batch, const float* __restrict__ xl2,
                           float* __restrict__ gden, float* __restrict__ gout, int N) {
  int tid = blockIdx.x * blockDim.x + threadIdx.x;
  if (tid >= N * 16) return;
  int n = tid >> 4, q = tid & 15;
  int b = batch[n];
  float a = __expf(logits2[n] - fdec(gmax[b]));
  if (q == 0) atomicAdd(&gden[b], a);
  int c0 = q * 4;
  float4 xv = *(const float4*)(xl2 + (size_t)n * 64 + c0);
  float* gp = gout + (size_t)b * 64 + c0;
  atomicAdd(gp + 0, a * xv.x);
  atomicAdd(gp + 1, a * xv.y);
  atomicAdd(gp + 2, a * xv.z);
  atomicAdd(gp + 3, a * xv.w);
}

// head: layernorm -> 2x silu dense 64x64 -> dot out_W ; one 64-thread block/graph
__global__ void k_final(const float* __restrict__ out2, const float* __restrict__ gln_g,
                        const float* __restrict__ gln_b, const float* __restrict__ postW,
                        const float* __restrict__ postb, const float* __restrict__ outW,
                        const float* __restrict__ outb, float* __restrict__ y) {
  __shared__ float s0[64];
  __shared__ float s1[64];
  int gi = blockIdx.x, j = threadIdx.x;
  float v = out2[(size_t)gi * 64 + j];
  s0[j] = v;
  __syncthreads();
  float mu = 0.f;
  for (int k = 0; k < 64; ++k) mu += s0[k];
  mu *= (1.f / 64.f);
  float var = 0.f;
  for (int k = 0; k < 64; ++k) { float d = s0[k] - mu; var += d * d; }
  var *= (1.f / 64.f);
  float z = (v - mu) * rsqrtf(var + 1e-5f) * gln_g[j] + gln_b[j];
  __syncthreads();
  s1[j] = z;
  __syncthreads();
  float a0 = postb[j];
  for (int k = 0; k < 64; ++k) a0 += s1[k] * postW[k * 64 + j];
  a0 = siluf(a0);
  __syncthreads();
  s0[j] = a0;
  __syncthreads();
  float a1 = postb[64 + j];
  for (int k = 0; k < 64; ++k) a1 += s0[k] * postW[4096 + k * 64 + j];
  a1 = siluf(a1);
  __syncthreads();
  s1[j] = a1;
  __syncthreads();
  if (j == 0) {
    float acc = outb[0];
    for (int k = 0; k < 64; ++k) acc += s1[k] * outW[k];
    y[gi] = acc;
  }
}

// ---------------------------------------------------------------------------
extern "C" void kernel_launch(void* const* d_in, const int* in_sizes, int n_in,
                              void* d_out, int out_size, void* d_ws, size_t ws_size,
                              hipStream_t stream) {
  (void)in_sizes; (void)n_in; (void)out_size; (void)ws_size;
  constexpr int N = 50000, E = 1000000, G = 512, H = 64, L = 3;

  const float* x          = (const float*)d_in[0];
  const float* edge_attr  = (const float*)d_in[1];
  const int*   eidx       = (const int*)d_in[2];
  const int*   batch      = (const int*)d_in[3];
  const float* pre_node_W = (const float*)d_in[4];
  const float* pre_node_b = (const float*)d_in[5];
  const float* pre_edge_W = (const float*)d_in[6];
  const float* pre_edge_b = (const float*)d_in[7];
  const float* ln_g    = (const float*)d_in[8];
  const float* ln_b    = (const float*)d_in[9];
  const float* gat_Wl  = (const float*)d_in[10];
  const float* gat_bl  = (const float*)d_in[11];
  const float* gat_Wr  = (const float*)d_in[12];
  const float* gat_br  = (const float*)d_in[13];
  const float* gat_We  = (const float*)d_in[14];
  const float* gat_be  = (const float*)d_in[15];
  const float* gat_att = (const float*)d_in[16];
  const float* gat_bias= (const float*)d_in[17];
  const float* glu_W   = (const float*)d_in[18];
  const float* glu_b   = (const float*)d_in[19];
  const float* gWl   = (const float*)d_in[20];
  const float* gbl   = (const float*)d_in[21];
  const float* gWr   = (const float*)d_in[22];
  const float* gbr   = (const float*)d_in[23];
  const float* gatt  = (const float*)d_in[24];
  const float* gbias = (const float*)d_in[25];
  const float* gglu_W = (const float*)d_in[26];
  const float* gglu_b = (const float*)d_in[27];
  const float* gln_g  = (const float*)d_in[28];
  const float* gln_b  = (const float*)d_in[29];
  const float* post_W = (const float*)d_in[30];
  const float* post_b = (const float*)d_in[31];
  const float* outW   = (const float*)d_in[32];
  const float* outb   = (const float*)d_in[33];
  const int* src = eidx;
  const int* dst = eidx + E;
  float* y = (float*)d_out;

  char* wsb = (char*)d_ws;
  size_t off = 0;
  auto alloc = [&](size_t bytes) -> void* {
    void* p = wsb + off;
    off += (bytes + 255) & ~(size_t)255;
    return p;
  };

  bf16_t*  ea_bf    = (bf16_t*)  alloc((size_t)E * H * sizeof(bf16_t));
  float*   h        = (float*)   alloc((size_t)N * H * sizeof(float));
  bf16_t*  hn_bf    = (bf16_t*)  alloc((size_t)N * H * sizeof(bf16_t));
  float*   xl       = (float*)   alloc((size_t)N * H * sizeof(float));
  float*   xr       = (float*)   alloc((size_t)N * H * sizeof(float));
  float*   gacc     = (float*)   alloc((size_t)N * H * sizeof(float));
  bf16_t*  g_bf     = (bf16_t*)  alloc((size_t)N * H * sizeof(bf16_t));
  bf16_t*  h_bf     = (bf16_t*)  alloc((size_t)N * H * sizeof(bf16_t));
  float*   logits   = (float*)   alloc((size_t)E * sizeof(float));
  unsigned* dmax    = (unsigned*)alloc((size_t)N * sizeof(unsigned));
  float*   den      = (float*)   alloc((size_t)N * sizeof(float));
  float*   pooled   = (float*)   alloc((size_t)G * H * sizeof(float));
  bf16_t*  pooled_bf= (bf16_t*)  alloc((size_t)G * H * sizeof(bf16_t));
  float*   xr2      = (float*)   alloc((size_t)G * H * sizeof(float));
  unsigned* gmax    = (unsigned*)alloc((size_t)G * sizeof(unsigned));
  float*   gden     = (float*)   alloc((size_t)G * sizeof(float));
  float*   gout     = (float*)   alloc((size_t)G * H * sizeof(float));
  bf16_t*  gout_bf  = (bf16_t*)  alloc((size_t)G * H * sizeof(bf16_t));
  float*   out2     = (float*)   alloc((size_t)G * H * sizeof(float));
  bf16_t*  Wl_bf    = (bf16_t*)  alloc((size_t)L * H * H * sizeof(bf16_t));
  bf16_t*  Wr_bf    = (bf16_t*)  alloc((size_t)L * H * H * sizeof(bf16_t));
  bf16_t*  We_bf    = (bf16_t*)  alloc((size_t)L * H * H * sizeof(bf16_t));
  bf16_t*  gluW_bf  = (bf16_t*)  alloc((size_t)L * 4 * H * H * sizeof(bf16_t));
  bf16_t*  gWl_bf   = (bf16_t*)  alloc((size_t)H * H * sizeof(bf16_t));
  bf16_t*  gWr_bf   = (bf16_t*)  alloc((size_t)H * H * sizeof(bf16_t));
  bf16_t*  gglu_bf  = (bf16_t*)  alloc((size_t)4 * H * H * sizeof(bf16_t));
  float*   xl2      = xl;       // reuse after GAT layers
  float*   logits2  = logits;   // reuse

  const int TB = 256;
  auto nblk = [](long total, int b) { return (int)((total + b - 1) / b); };

  // weight conversions: f32 [K,N] -> bf16 transposed [N,K]
  k_cvt_t<<<nblk(L * H * H, TB), TB, 0, stream>>>(gat_Wl, Wl_bf, H, H, L * H * H);
  k_cvt_t<<<nblk(L * H * H, TB), TB, 0, stream>>>(gat_Wr, Wr_bf, H, H, L * H * H);
  k_cvt_t<<<nblk(L * H * H, TB), TB, 0, stream>>>(gat_We, We_bf, H, H, L * H * H);
  k_cvt_t<<<nblk(L * 4 * H * H, TB), TB, 0, stream>>>(glu_W, gluW_bf, 2 * H, 2 * H, L * 4 * H * H);
  k_cvt_t<<<nblk(H * H, TB), TB, 0, stream>>>(gWl, gWl_bf, H, H, H * H);
  k_cvt_t<<<nblk(H * H, TB), TB, 0, stream>>>(gWr, gWr_bf, H, H, H * H);
  k_cvt_t<<<nblk(4 * H * H, TB), TB, 0, stream>>>(gglu_W, gglu_bf, 2 * H, 2 * H, 4 * H * H);

  // front-end dense layers
  k_pre_node<<<nblk((long)N * H, TB), TB, 0, stream>>>(x, pre_node_W, pre_node_b, h, N * H);
  k_pre_edge<<<nblk((long)E * H, TB), TB, 0, stream>>>(edge_attr, pre_edge_W, pre_edge_b, ea_bf, E * H);

  const int mt = N / 16;   // 3125 node tiles
  const int et = E / 16;   // 62500 edge tiles
  const int gt = G / 16;   // 32 graph tiles

  for (int l = 0; l < L; ++l) {
    k_layernorm_bf<<<nblk(N, TB), TB, 0, stream>>>(h, ln_g + l * H, ln_b + l * H, hn_bf, N);
    k_gemm64_dual<<<(mt + 7) / 8, TB, 0, stream>>>(hn_bf,
                                                   Wl_bf + l * H * H, gat_bl + l * H, xl,
                                                   Wr_bf + l * H * H, gat_br + l * H, xr, mt);
    k_init_seg<<<nblk((long)N * H, TB), TB, 0, stream>>>(dmax, den, gacc, N);
    k_edge_logit<<<(et + 7) / 8, TB, 0, stream>>>(ea_bf, We_bf + l * H * H, gat_be + l * H,
                                                  gat_att + l * H, xl, xr, src, dst,
                                                  logits, dmax, et);
    k_edge_scatter<<<nblk((long)E * 16, TB), TB, 0, stream>>>(logits, dmax, src, dst, xl,
                                                              den, gacc, E);
    k_g_final<<<nblk((long)N * H, TB), TB, 0, stream>>>(gacc, den, gat_bias + l * H, g_bf, N * H);
    k_glu<<<(mt + 7) / 8, TB, 0, stream>>>(hn_bf, g_bf, gluW_bf + l * 4 * H * H,
                                           glu_b + l * 2 * H, h, mt);
  }

  // global super-node attention
  k_zero<<<nblk(G * H, TB), TB, 0, stream>>>(pooled, G * H);
  k_pool_sum<<<nblk((long)N * H, TB), TB, 0, stream>>>(h, batch, pooled, N * H);
  k_pool_relu<<<nblk(G * H, TB), TB, 0, stream>>>(pooled, pooled_bf, G * H);
  k_cvt_bf16<<<nblk((long)N * H, TB), TB, 0, stream>>>(h, h_bf, N * H);
  k_gemm64<<<(mt + 7) / 8, TB, 0, stream>>>(h_bf, gWl_bf, gbl, xl2, mt);
  k_gemm64<<<(gt + 7) / 8, TB, 0, stream>>>(pooled_bf, gWr_bf, gbr, xr2, gt);
  k_init_seg<<<nblk(G * H, TB), TB, 0, stream>>>(gmax, gden, gout, G);
  k_gnode_logit<<<nblk(N, TB), TB, 0, stream>>>(xl2, xr2, gatt, batch, logits2, gmax, N);
  k_gscatter<<<nblk((long)N * 16, TB), TB, 0, stream>>>(logits2, gmax, batch, xl2, gden, gout, N);
  k_g_final<<<nblk(G * H, TB), TB, 0, stream>>>(gout, gden, gbias, gout_bf, G * H);
  k_glu<<<(gt + 7) / 8, TB, 0, stream>>>(pooled_bf, gout_bf, gglu_bf, gglu_b, out2, gt);

  // head
  k_final<<<G, H, 0, stream>>>(out2, gln_g, gln_b, post_W, post_b, outW, outb, y);
}
